// PolyaSimpleKTNetV5_26147760898074
// MI455X (gfx1250) — matrix operand order repaired and checked by reading, not compiled, not run on hardware
//
#include <hip/hip_runtime.h>
#include <hip/hip_bf16.h>

// ---------------- types ----------------
typedef __bf16 bf16x16 __attribute__((ext_vector_type(16)));
typedef float  f32x8   __attribute__((ext_vector_type(8)));
typedef unsigned int u32x4 __attribute__((ext_vector_type(4)));
typedef int i32x4 __attribute__((ext_vector_type(4)));
typedef int i32x8 __attribute__((ext_vector_type(8)));

union AF { uint4 q[2]; bf16x16 v; };

static __device__ __forceinline__ unsigned short f2bf(float f) {
    unsigned int u = __float_as_uint(f);
    unsigned int r = (u + 0x7FFFu + ((u >> 16) & 1u)) >> 16;
    return (unsigned short)r;
}

static __device__ __forceinline__ float pe_val(int s, int d) {
    const float kk = -9.210340371976184f / 512.0f;   // -ln(10000)/D
    float fr = __expf(kk * (float)(d & ~1));
    float a = (float)s * fr;
    return (d & 1) ? __cosf(a) : __sinf(a);
}

// LDS byte address (low 32 bits of the flat address are the LDS offset on CDNA5)
static __device__ __forceinline__ unsigned lds_addr32(const void* p) {
    return (unsigned)(unsigned long long)p;
}

// ---------------- Tensor Data Mover: 2-D strided bf16 tile -> LDS ----------------
static __device__ __forceinline__ void tdm_load_tile_bf16(
    unsigned lds_off, const void* gaddr, int tile_k, int tile_rows, long row_stride_elems)
{
    unsigned long long ga = (unsigned long long)gaddr;
    u32x4 g0;
    g0.x = 1u;                                   // count=1, user descriptor
    g0.y = lds_off;                              // lds_addr
    g0.z = (unsigned)(ga & 0xFFFFFFFFu);         // global_addr[31:0]
    g0.w = (unsigned)((ga >> 32) & 0x01FFFFFFu)  // global_addr[56:32]
         | (2u << 30);                           // type = 2 ("image")
    const unsigned td0 = 0x40000000u;            // huge tensor_dim0 (no OOB clipping)
    const unsigned td1 = 0x40000000u;            // huge tensor_dim1
    unsigned long long st = (unsigned long long)row_stride_elems;  // data_size units
    i32x8 g1;
    g1[0] = (int)(1u << 16);                     // wg_mask=0, data_size=1 (2B)
    g1[1] = (int)((td0 & 0xFFFFu) << 16);        // atomic_addr=0 | tensor_dim0 lo16
    g1[2] = (int)((td0 >> 16) | ((td1 & 0xFFFFu) << 16));
    g1[3] = (int)((td1 >> 16) | ((unsigned)tile_k << 16));   // td1 hi | tile_dim0
    g1[4] = (int)(unsigned)tile_rows;            // tile_dim1 | tile_dim2=0
    g1[5] = (int)(unsigned)(st & 0xFFFFFFFFu);   // tensor_dim0_stride lo32
    g1[6] = (int)(unsigned)((st >> 32) & 0xFFFFu);
    g1[7] = 0;
    i32x4 z4 = {0, 0, 0, 0};
#if __clang_major__ >= 23
    i32x8 z8 = {0, 0, 0, 0, 0, 0, 0, 0};
    __builtin_amdgcn_tensor_load_to_lds(g0, g1, z4, z4, z8, 0);
#else
    __builtin_amdgcn_tensor_load_to_lds(g0, g1, z4, z4, 0);
#endif
}

// async 16B global -> LDS copy (per lane), tracked by ASYNCcnt
static __device__ __forceinline__ void async_b128_to_lds(unsigned lds_off, const void* gp) {
    asm volatile("global_load_async_to_lds_b128 %0, %1, off"
                 :: "v"(lds_off), "v"(gp) : "memory");
}
static __device__ __forceinline__ void wait_asynccnt0() {
    asm volatile("s_wait_asynccnt 0x0" ::: "memory");
}
static __device__ __forceinline__ void wait_asynccnt8() {
    asm volatile("s_wait_asynccnt 0x8" ::: "memory");
}

// ---------------- f32 -> bf16 convert (plain) ----------------
__global__ void cvt_bf16_kernel(const float* __restrict__ in, unsigned short* __restrict__ out, long n) {
    long i = (long)blockIdx.x * 256 + threadIdx.x;
    if (i < n) out[i] = f2bf(in[i]);
}

// f32 (Kd x Nd, row-major) -> bf16 transposed (Nd x Kd), batched on blockIdx.y
__global__ void cvt_bf16_t_kernel(const float* __restrict__ in, unsigned short* __restrict__ out,
                                  int Kd, int Nd) {
    long base = (long)blockIdx.y * Kd * Nd;
    long idx = (long)blockIdx.x * 256 + threadIdx.x;
    if (idx < (long)Kd * Nd) {
        int k = (int)(idx / Nd);
        int n = (int)(idx % Nd);
        out[base + (long)n * Kd + k] = f2bf(in[base + idx]);
    }
}

__global__ void copy_f32_kernel(const float* __restrict__ in, float* __restrict__ out, long n) {
    long i = (long)blockIdx.x * 256 + threadIdx.x;
    if (i < n) out[i] = in[i];
}

// ---------------- generic bf16 WMMA GEMM (double-buffered TDM pipeline) ----------------
// C[M,N] = A[M,K] * B[K,N] + bias.  A: bf16 row-major (lda).
// B passed TRANSPOSED: Bt is (N x K) row-major, ldb = K-stride. Batched via blockIdx.z.
#define GF_RELU 1
#define GF_OBF16 2
#define GF_VT 4   // store output as (b, h, dk, s) bf16 per-head transposed (H=8, DK=64)

__global__ void __launch_bounds__(256)
gemm_bf16_wmma(const unsigned short* __restrict__ A, long sAb, int lda,
               const unsigned short* __restrict__ Bt, long sBb, int ldb,
               const float* __restrict__ bias, long sBiasb,
               void* __restrict__ Cout, long sCb, int ldc,
               int K, int flags, int vtS)
{
    __shared__ __align__(16) unsigned short As[2][128][32];
    __shared__ __align__(16) unsigned short Bs[2][128][32];   // [n][k]

    const int tid  = threadIdx.x;
    const int lane = tid & 31;
    const int wid  = tid >> 5;
    const int wm   = wid >> 2;   // 0..1
    const int wn   = wid & 3;    // 0..3
    const int bm0  = blockIdx.y * 128;
    const int bn0  = blockIdx.x * 128;
    const int bz   = blockIdx.z;

    A  += (long)bz * sAb;
    Bt += (long)bz * sBb;
    const float* biasp = bias + (long)bz * sBiasb;

    f32x8 acc[4][2] = {};

    const int lm = lane & 15;
    const int hi = lane >> 4;
    const int kb = hi * 8;
    const int klo = hi * 16;
    const unsigned asOff[2] = { lds_addr32(&As[0][0][0]), lds_addr32(&As[1][0][0]) };
    const unsigned bsOff[2] = { lds_addr32(&Bs[0][0][0]), lds_addr32(&Bs[1][0][0]) };

    const int nk = K >> 5;
    // prologue: DMA tile 0 into buffer 0
    if (wid == 0) {
        tdm_load_tile_bf16(asOff[0], &A[(long)bm0 * lda], 32, 128, lda);
        tdm_load_tile_bf16(bsOff[0], &Bt[(long)bn0 * ldb], 32, 128, ldb);
    }

    for (int t = 0; t < nk; ++t) {
        const int buf = t & 1;
        if (wid == 0) {
            if (t + 1 < nk) {
                // DMA next tile into the spare buffer, keep it in flight
                int k1 = (t + 1) << 5;
                tdm_load_tile_bf16(asOff[buf ^ 1], &A[(long)bm0 * lda + k1], 32, 128, lda);
                tdm_load_tile_bf16(bsOff[buf ^ 1], &Bt[(long)bn0 * ldb + k1], 32, 128, ldb);
                __builtin_amdgcn_s_wait_tensorcnt((short)2);   // current tile landed
            } else {
                __builtin_amdgcn_s_wait_tensorcnt((short)0);
            }
        }
        __syncthreads();

        const unsigned short (*Ap)[32] = As[buf];
        const unsigned short (*Bp)[32] = Bs[buf];
        AF afr[4], bfr[2];
        #pragma unroll
        for (int mt = 0; mt < 4; mt++) {
            int row = wm * 64 + mt * 16 + lm;
            afr[mt].q[0] = *reinterpret_cast<const uint4*>(&Ap[row][kb]);
            afr[mt].q[1] = *reinterpret_cast<const uint4*>(&Ap[row][16 + kb]);
        }
        #pragma unroll
        for (int nt = 0; nt < 2; nt++) {
            int col = wn * 32 + nt * 16 + lm;
            bfr[nt].q[0] = *reinterpret_cast<const uint4*>(&Bp[col][kb]);
            bfr[nt].q[1] = *reinterpret_cast<const uint4*>(&Bp[col][16 + kb]);
        }
        #pragma unroll
        for (int mt = 0; mt < 4; mt++)
            #pragma unroll
            for (int nt = 0; nt < 2; nt++)
                acc[mt][nt] = __builtin_amdgcn_wmma_f32_16x16x32_bf16(
                    false, afr[mt].v, false, bfr[nt].v, (short)0, acc[mt][nt], false, false);
        __syncthreads();   // everyone done reading buf before it is re-filled
    }

    const bool relu = (flags & GF_RELU) != 0;
    const bool obf  = (flags & GF_OBF16) != 0;
    const bool vt   = (flags & GF_VT) != 0;
    #pragma unroll
    for (int mt = 0; mt < 4; mt++) {
        #pragma unroll
        for (int nt = 0; nt < 2; nt++) {
            #pragma unroll
            for (int r = 0; r < 8; r++) {
                int m = bm0 + wm * 64 + mt * 16 + r + 8 * hi;
                int n = bn0 + wn * 32 + nt * 16 + lm;
                float v = acc[mt][nt][r] + biasp[n];
                if (relu) v = fmaxf(v, 0.f);
                if (vt) {
                    int bb = m / vtS, s = m % vtS;
                    int hh = n >> 6, dk = n & 63;
                    long idx = (((long)bb * 8 + hh) * 64 + dk) * vtS + s;
                    ((unsigned short*)Cout)[idx] = f2bf(v);
                } else {
                    long idx = (long)bz * sCb + (long)m * ldc + n;
                    if (obf) ((unsigned short*)Cout)[idx] = f2bf(v);
                    else     ((float*)Cout)[idx] = v;
                }
            }
        }
    }
}

// ---------------- MoE gate + top2 + softmax + combine + LN + PE ----------------
__global__ void __launch_bounds__(256)
moe_gate_combine_ln(const float* __restrict__ summary,
                    const float* __restrict__ m_seq,
                    const float* __restrict__ gate_w,
                    const float* __restrict__ gate_b,
                    const float* __restrict__ experts,
                    const float* __restrict__ ln_g, const float* __restrict__ ln_b,
                    float* __restrict__ x_f32, unsigned short* __restrict__ x_bf16,
                    int S, int R)
{
    const int row = blockIdx.x;
    const int tid = threadIdx.x;
    __shared__ float red[256];
    __shared__ float gw[4];
    __shared__ float st[2];

    float part[4] = {0.f, 0.f, 0.f, 0.f};
    for (int c = tid; c < 576; c += 256) {
        float v = (c < 512) ? summary[(long)row * 512 + c] : m_seq[(long)row * 64 + (c - 512)];
        #pragma unroll
        for (int k = 0; k < 4; k++) part[k] += v * gate_w[c * 4 + k];
    }
    #pragma unroll
    for (int k = 0; k < 4; k++) {
        red[tid] = part[k];
        __syncthreads();
        for (int off = 128; off > 0; off >>= 1) {
            if (tid < off) red[tid] += red[tid + off];
            __syncthreads();
        }
        if (tid == 0) gw[k] = red[0] + gate_b[k];
        __syncthreads();
    }
    if (tid == 0) {
        float ls[4] = {gw[0], gw[1], gw[2], gw[3]};
        float mx = fmaxf(fmaxf(ls[0], ls[1]), fmaxf(ls[2], ls[3]));
        float s2 = -3.0e38f; bool skipped = false;
        for (int k = 0; k < 4; k++) {
            if (!skipped && ls[k] == mx) { skipped = true; continue; }
            s2 = fmaxf(s2, ls[k]);
        }
        float se = 0.f; float w[4];
        for (int k = 0; k < 4; k++) {
            w[k] = (ls[k] >= s2) ? __expf(ls[k] - mx) : 0.f;
            se += w[k];
        }
        for (int k = 0; k < 4; k++) gw[k] = w[k] / se;
    }
    __syncthreads();
    float g0 = gw[0], g1 = gw[1], g2 = gw[2], g3 = gw[3];
    long es = (long)R * 512;
    float comb[2]; float lsum = 0.f, lsq = 0.f;
    #pragma unroll
    for (int i = 0; i < 2; i++) {
        int d = tid + i * 256;
        long off = (long)row * 512 + d;
        float c = g0 * experts[off] + g1 * experts[es + off] +
                  g2 * experts[2 * es + off] + g3 * experts[3 * es + off];
        comb[i] = c; lsum += c; lsq += c * c;
    }
    red[tid] = lsum; __syncthreads();
    for (int off = 128; off > 0; off >>= 1) { if (tid < off) red[tid] += red[tid + off]; __syncthreads(); }
    if (tid == 0) st[0] = red[0] / 512.f;
    __syncthreads();
    red[tid] = lsq; __syncthreads();
    for (int off = 128; off > 0; off >>= 1) { if (tid < off) red[tid] += red[tid + off]; __syncthreads(); }
    if (tid == 0) { float mu = st[0]; st[1] = rsqrtf(red[0] / 512.f - mu * mu + 1e-5f); }
    __syncthreads();
    float mu = st[0], inv = st[1];
    int s = row % S;
    #pragma unroll
    for (int i = 0; i < 2; i++) {
        int d = tid + i * 256;
        float nv = (comb[i] - mu) * inv * ln_g[d] + ln_b[d] + pe_val(s, d);
        x_f32[(long)row * 512 + d] = nv;
        x_bf16[(long)row * 512 + d] = f2bf(nv);
    }
}

// ---------------- y = qa_embed + pe (bf16) ----------------
__global__ void __launch_bounds__(256)
make_y_kernel(const float* __restrict__ qa, unsigned short* __restrict__ y, int S)
{
    int row = blockIdx.x;
    int tid = threadIdx.x;
    int s = row % S;
    #pragma unroll
    for (int i = 0; i < 2; i++) {
        int d = tid + i * 256;
        y[(long)row * 512 + d] = f2bf(qa[(long)row * 512 + d] + pe_val(s, d));
    }
}

// ---------------- residual + LayerNorm ----------------
__global__ void __launch_bounds__(256)
resid_ln_kernel(float* __restrict__ x, const float* __restrict__ g,
                const float* __restrict__ gamma, const float* __restrict__ beta,
                unsigned short* __restrict__ xbf)
{
    const int row = blockIdx.x;
    const int tid = threadIdx.x;
    __shared__ float red[256];
    __shared__ float st[2];
    float v[2]; float s = 0.f, q = 0.f;
    #pragma unroll
    for (int i = 0; i < 2; i++) {
        int d = tid + i * 256;
        float t = x[(long)row * 512 + d] + g[(long)row * 512 + d];
        v[i] = t; s += t; q += t * t;
    }
    red[tid] = s; __syncthreads();
    for (int off = 128; off > 0; off >>= 1) { if (tid < off) red[tid] += red[tid + off]; __syncthreads(); }
    if (tid == 0) st[0] = red[0] / 512.f;
    __syncthreads();
    red[tid] = q; __syncthreads();
    for (int off = 128; off > 0; off >>= 1) { if (tid < off) red[tid] += red[tid + off]; __syncthreads(); }
    if (tid == 0) { float mu = st[0]; st[1] = rsqrtf(red[0] / 512.f - mu * mu + 1e-5f); }
    __syncthreads();
    float mu = st[0], inv = st[1];
    #pragma unroll
    for (int i = 0; i < 2; i++) {
        int d = tid + i * 256;
        float nv = (v[i] - mu) * inv * gamma[d] + beta[d];
        x[(long)row * 512 + d] = nv;
        xbf[(long)row * 512 + d] = f2bf(nv);
    }
}

// ---------------- flash attention (double-buffered async K/V pipeline) ----------------
// kh: (b*S, D) bf16.  vhT: (b, h, dk, s) bf16.  grid (S/64, H, B), block 128 (4 waves).
__global__ void __launch_bounds__(128)
attn_flash_kernel(const unsigned short* __restrict__ kh,
                  const unsigned short* __restrict__ vhT,
                  unsigned short* __restrict__ ctx,
                  int S, int D, int H)
{
    const int ib = blockIdx.x;
    const int h  = blockIdx.y;
    const int b  = blockIdx.z;
    const int tid  = threadIdx.x;
    const int w    = tid >> 5;
    const int lane = tid & 31;
    const int lm = lane & 15;
    const int hi = lane >> 4;
    const int kb = hi * 8;

    __shared__ __align__(16) unsigned short Ks[2][64][64];   // [j][dk]
    __shared__ __align__(16) unsigned short Vt[2][64][64];   // [dk][j]
    __shared__ __align__(16) unsigned short Pb[4][16][64];

    const long rowbase = (long)b * S;
    const int  colh = h * 64;
    const int  ibase = ib * 64;
    const int  qrow = ibase + w * 16 + lm;
    const long vbase = ((long)b * H + h) * 64;

    AF qa0, qa1;
    {
        const unsigned short* qp = kh + (rowbase + qrow) * D + colh;
        qa0.q[0] = *reinterpret_cast<const uint4*>(qp + kb);
        qa0.q[1] = *reinterpret_cast<const uint4*>(qp + 16 + kb);
        qa1.q[0] = *reinterpret_cast<const uint4*>(qp + 32 + kb);
        qa1.q[1] = *reinterpret_cast<const uint4*>(qp + 48 + kb);
    }

    // per-wave async stage of one 64x64 K tile + one 64x64 V tile (8 instructions/wave)
    auto stage = [&](int jbase, int buf) {
        #pragma unroll
        for (int i = 0; i < 4; i++) {
            int c = tid + i * 128;
            int r0 = c >> 3;
            int e0 = (c & 7) * 8;
            async_b128_to_lds(lds_addr32(&Ks[buf][r0][e0]),
                              &kh[(rowbase + jbase + r0) * D + colh + e0]);
            async_b128_to_lds(lds_addr32(&Vt[buf][r0][e0]),
                              &vhT[(vbase + r0) * S + jbase + e0]);
        }
    };

    f32x8 o[4] = {};
    float mrow[8], lrow[8];
    #pragma unroll
    for (int r = 0; r < 8; r++) { mrow[r] = -3.0e38f; lrow[r] = 0.f; }

    stage(0, 0);   // prologue: tile 0 in flight

    for (int jb = 0; jb <= ib; ++jb) {
        const int jbase = jb * 64;
        const int buf = jb & 1;
        if (jb < ib) {               // prefetch next tile into spare buffer
            stage(jbase + 64, buf ^ 1);
            wait_asynccnt8();        // current tile complete, next 8 still in flight
        } else {
            wait_asynccnt0();
        }
        __syncthreads();

        const unsigned short (*Kp)[64] = Ks[buf];
        const unsigned short (*Vp)[64] = Vt[buf];

        // S tile 16x64 per wave, DK=64 in two K=32 WMMA steps
        f32x8 sc[4] = {};
        const int klo = hi * 16;
        #pragma unroll
        for (int jt = 0; jt < 4; jt++) {
            AF bf0, bf1;
            int col = jt * 16 + lm;
            bf0.q[0] = *reinterpret_cast<const uint4*>(&Kp[col][kb]);
            bf0.q[1] = *reinterpret_cast<const uint4*>(&Kp[col][16 + kb]);
            bf1.q[0] = *reinterpret_cast<const uint4*>(&Kp[col][32 + kb]);
            bf1.q[1] = *reinterpret_cast<const uint4*>(&Kp[col][48 + kb]);
            sc[jt] = __builtin_amdgcn_wmma_f32_16x16x32_bf16(false, qa0.v, false, bf0.v, (short)0, sc[jt], false, false);
            sc[jt] = __builtin_amdgcn_wmma_f32_16x16x32_bf16(false, qa1.v, false, bf1.v, (short)0, sc[jt], false, false);
        }
        const bool diag = (jb == ib);
        #pragma unroll
        for (int jt = 0; jt < 4; jt++) {
            #pragma unroll
            for (int r = 0; r < 8; r++) {
                float v = sc[jt][r] * 0.125f;
                if (diag) {
                    int gi = ibase + w * 16 + r + 8 * hi;
                    int gj = jbase + jt * 16 + lm;
                    if (gj >= gi) v = -1e30f;
                }
                sc[jt][r] = v;
            }
        }
        // online softmax
        #pragma unroll
        for (int r = 0; r < 8; r++) {
            float cm = fmaxf(fmaxf(sc[0][r], sc[1][r]), fmaxf(sc[2][r], sc[3][r]));
            #pragma unroll
            for (int msk = 1; msk < 16; msk <<= 1) cm = fmaxf(cm, __shfl_xor(cm, msk, 32));
            float mn = fmaxf(mrow[r], cm);
            float sc_old = __expf(mrow[r] - mn);
            mrow[r] = mn;
            float rs = 0.f;
            #pragma unroll
            for (int jt = 0; jt < 4; jt++) {
                float p = __expf(sc[jt][r] - mn);
                sc[jt][r] = p;
                rs += p;
            }
            #pragma unroll
            for (int msk = 1; msk < 16; msk <<= 1) rs += __shfl_xor(rs, msk, 32);
            lrow[r] = lrow[r] * sc_old + rs;
            #pragma unroll
            for (int nt = 0; nt < 4; nt++) o[nt][r] *= sc_old;
        }
        // P (C layout) -> LDS row-major -> A-fragment layout
        #pragma unroll
        for (int jt = 0; jt < 4; jt++)
            #pragma unroll
            for (int r = 0; r < 8; r++)
                Pb[w][r + 8 * hi][jt * 16 + lm] = f2bf(sc[jt][r]);
        __syncthreads();
        // O += P @ V
        #pragma unroll
        for (int kc = 0; kc < 2; kc++) {
            AF pa;
            pa.q[0] = *reinterpret_cast<const uint4*>(&Pb[w][lm][kc * 32 + kb]);
            pa.q[1] = *reinterpret_cast<const uint4*>(&Pb[w][lm][kc * 32 + 16 + kb]);
            #pragma unroll
            for (int nt = 0; nt < 4; nt++) {
                AF vbf;
                int dk = nt * 16 + lm;
                int kl2 = kc * 32 + hi * 8;
                vbf.q[0] = *reinterpret_cast<const uint4*>(&Vp[dk][kl2]);
                vbf.q[1] = *reinterpret_cast<const uint4*>(&Vp[dk][kl2 + 16]);
                o[nt] = __builtin_amdgcn_wmma_f32_16x16x32_bf16(false, pa.v, false, vbf.v, (short)0, o[nt], false, false);
            }
        }
        __syncthreads();   // done reading buf before it is re-staged
    }

    #pragma unroll
    for (int nt = 0; nt < 4; nt++) {
        #pragma unroll
        for (int r = 0; r < 8; r++) {
            int gi = ibase + w * 16 + r + 8 * hi;
            float v = o[nt][r] / fmaxf(lrow[r], 1e-20f);
            if (gi == 0) v = 0.f;
            ctx[(rowbase + gi) * D + colh + nt * 16 + lm] = f2bf(v);
        }
    }
}

// ---------------- host launcher ----------------
extern "C" void kernel_launch(void* const* d_in, const int* in_sizes, int n_in,
                              void* d_out, int out_size, void* d_ws, size_t ws_size,
                              hipStream_t stream) {
    (void)in_sizes; (void)n_in; (void)out_size; (void)ws_size;

    const int Bb = 4, S = 1024, D = 512, H = 8, FF = 2048, L = 4, IN = 768, NS = 4;
    const int R = Bb * S;
    const long RD = (long)R * D;

    const float* q_raw   = (const float*)d_in[0];
    const float* m_seq   = (const float*)d_in[1];
    const float* qa_emb  = (const float*)d_in[2];
    const float* exp_w1  = (const float*)d_in[3];
    const float* exp_b1  = (const float*)d_in[4];
    const float* exp_w2  = (const float*)d_in[5];
    const float* exp_b2  = (const float*)d_in[6];
    const float* adapt_w = (const float*)d_in[7];
    const float* adapt_b = (const float*)d_in[8];
    const float* gate_w  = (const float*)d_in[9];
    const float* gate_b  = (const float*)d_in[10];
    const float* ln_g    = (const float*)d_in[11];
    const float* ln_b    = (const float*)d_in[12];
    const float* kw      = (const float*)d_in[13];
    const float* kbias   = (const float*)d_in[14];
    const float* vw      = (const float*)d_in[15];
    const float* vbias   = (const float*)d_in[16];
    const float* ow      = (const float*)d_in[17];
    const float* obias   = (const float*)d_in[18];
    const float* ln1_g   = (const float*)d_in[19];
    const float* ln1_b   = (const float*)d_in[20];
    const float* ff_w1   = (const float*)d_in[21];
    const float* ff_b1   = (const float*)d_in[22];
    const float* ff_w2   = (const float*)d_in[23];
    const float* ff_b2   = (const float*)d_in[24];
    const float* ln2_g   = (const float*)d_in[25];
    const float* ln2_b   = (const float*)d_in[26];
    float* out = (float*)d_out;

    char* ws = (char*)d_ws;
    size_t off = 0;
    auto carve = [&](size_t bytes) -> char* {
        char* p = ws + off;
        off += (bytes + 255) & ~(size_t)255;
        return p;
    };
    unsigned short* qraw_bf = (unsigned short*)carve((size_t)R * NS * IN * 2);
    unsigned short* h1_bf   = (unsigned short*)carve((size_t)R * NS * IN * 2);
    float*          exp_f   = (float*)carve((size_t)NS * R * D * 4);
    float*          summ_f  = (float*)carve((size_t)R * D * 4);
    float*          x_f     = (float*)carve((size_t)R * D * 4);
    unsigned short* x_bf    = (unsigned short*)carve((size_t)R * D * 2);
    unsigned short* y_bf    = (unsigned short*)carve((size_t)R * D * 2);
    unsigned short* kh_bf   = (unsigned short*)carve((size_t)R * D * 2);
    unsigned short* vh_bf   = (unsigned short*)carve((size_t)R * D * 2); // vhT (b,h,dk,s)
    unsigned short* ctx_bf  = (unsigned short*)carve((size_t)R * D * 2);
    float*          tmp_f   = (float*)carve((size_t)R * D * 4);
    unsigned short* ffh_bf  = (unsigned short*)carve((size_t)R * FF * 2);
    unsigned short* we1_bf  = (unsigned short*)carve((size_t)NS * IN * IN * 2);
    unsigned short* we2_bf  = (unsigned short*)carve((size_t)NS * IN * D * 2);
    unsigned short* wad_bf  = (unsigned short*)carve((size_t)NS * IN * D * 2);
    unsigned short* wk_bf   = (unsigned short*)carve((size_t)L * D * D * 2);
    unsigned short* wv_bf   = (unsigned short*)carve((size_t)L * D * D * 2);
    unsigned short* wo_bf   = (unsigned short*)carve((size_t)L * D * D * 2);
    unsigned short* wf1_bf  = (unsigned short*)carve((size_t)L * D * FF * 2);
    unsigned short* wf2_bf  = (unsigned short*)carve((size_t)L * FF * D * 2);

    {
        long n = (long)R * NS * IN;
        cvt_bf16_kernel<<<(int)((n + 255) / 256), 256, 0, stream>>>(q_raw, qraw_bf, n);
    }
    auto cvtT = [&](const float* src, unsigned short* dst, int Kd, int Nd, int nb) {
        long per = (long)Kd * Nd;
        cvt_bf16_t_kernel<<<dim3((unsigned)((per + 255) / 256), nb), 256, 0, stream>>>(src, dst, Kd, Nd);
    };
    cvtT(exp_w1,  we1_bf, IN, IN, NS);
    cvtT(exp_w2,  we2_bf, IN, D, NS);
    cvtT(adapt_w, wad_bf, NS * IN, D, 1);
    cvtT(kw,      wk_bf,  D, D, L);
    cvtT(vw,      wv_bf,  D, D, L);
    cvtT(ow,      wo_bf,  D, D, L);
    cvtT(ff_w1,   wf1_bf, D, FF, L);
    cvtT(ff_w2,   wf2_bf, FF, D, L);

    // expert GEMM1: per-expert (M=4096, K=768, N=768), relu, bf16 out
    gemm_bf16_wmma<<<dim3(IN / 128, R / 128, NS), 256, 0, stream>>>(
        qraw_bf, (long)IN, NS * IN,
        we1_bf, (long)IN * IN, IN,
        exp_b1, (long)IN,
        h1_bf, (long)IN, NS * IN, IN, GF_RELU | GF_OBF16, S);

    // expert GEMM2: per-expert (M=4096, K=768, N=512), f32 out
    gemm_bf16_wmma<<<dim3(D / 128, R / 128, NS), 256, 0, stream>>>(
        h1_bf, (long)IN, NS * IN,
        we2_bf, (long)IN * D, IN,
        exp_b2, (long)D,
        exp_f, (long)R * D, D, IN, 0, S);

    // adapter: (M=4096, K=3072, N=512), relu, f32 out
    gemm_bf16_wmma<<<dim3(D / 128, R / 128, 1), 256, 0, stream>>>(
        qraw_bf, 0, NS * IN,
        wad_bf, 0, NS * IN,
        adapt_b, 0,
        summ_f, 0, D, NS * IN, GF_RELU, S);

    make_y_kernel<<<R, 256, 0, stream>>>(qa_emb, y_bf, S);

    moe_gate_combine_ln<<<R, 256, 0, stream>>>(
        summ_f, m_seq, gate_w, gate_b, exp_f, ln_g, ln_b, x_f, x_bf, S, R);

    for (int i = 0; i < L; i++) {
        gemm_bf16_wmma<<<dim3(D / 128, R / 128, 1), 256, 0, stream>>>(
            x_bf, 0, D, wk_bf + (long)i * D * D, 0, D,
            kbias + (long)i * D, 0, kh_bf, 0, D, D, GF_OBF16, S);
        gemm_bf16_wmma<<<dim3(D / 128, R / 128, 1), 256, 0, stream>>>(
            y_bf, 0, D, wv_bf + (long)i * D * D, 0, D,
            vbias + (long)i * D, 0, vh_bf, 0, D, D, GF_OBF16 | GF_VT, S);
        attn_flash_kernel<<<dim3(S / 64, H, Bb), 128, 0, stream>>>(kh_bf, vh_bf, ctx_bf, S, D, H);
        gemm_bf16_wmma<<<dim3(D / 128, R / 128, 1), 256, 0, stream>>>(
            ctx_bf, 0, D, wo_bf + (long)i * D * D, 0, D,
            obias + (long)i * D, 0, tmp_f, 0, D, D, 0, S);
        resid_ln_kernel<<<R, 256, 0, stream>>>(x_f, tmp_f,
            ln1_g + (long)i * D, ln1_b + (long)i * D, x_bf);
        gemm_bf16_wmma<<<dim3(FF / 128, R / 128, 1), 256, 0, stream>>>(
            x_bf, 0, D, wf1_bf + (long)i * D * FF, 0, D,
            ff_b1 + (long)i * FF, 0, ffh_bf, 0, FF, D, GF_RELU | GF_OBF16, S);
        gemm_bf16_wmma<<<dim3(D / 128, R / 128, 1), 256, 0, stream>>>(
            ffh_bf, 0, FF, wf2_bf + (long)i * FF * D, 0, FF,
            ff_b2 + (long)i * D, 0, tmp_f, 0, D, FF, 0, S);
        resid_ln_kernel<<<R, 256, 0, stream>>>(x_f, tmp_f,
            ln2_g + (long)i * D, ln2_b + (long)i * D, x_bf);
    }

    copy_f32_kernel<<<(int)((RD + 255) / 256), 256, 0, stream>>>(x_f, out, RD);
}
